// _GATConvGlobalGlobalAttention_56951266345490
// MI455X (gfx1250) — matrix operand order repaired and checked by reading, not compiled, not run on hardware
//
#include <hip/hip_runtime.h>
#include <hip/hip_bf16.h>

typedef __attribute__((ext_vector_type(16))) __bf16 v16bf;
typedef __attribute__((ext_vector_type(8)))  float  v8f;

#define NEG_SLOPE_F 0.2f

__device__ __forceinline__ float leaky(float x) { return x >= 0.f ? x : NEG_SLOPE_F * x; }

// monotonic float<->uint mapping for atomicMax on floats
__device__ __forceinline__ unsigned enc_f(float f) {
    unsigned u = __float_as_uint(f);
    return (u & 0x80000000u) ? ~u : (u | 0x80000000u);
}
__device__ __forceinline__ float dec_f(unsigned k) {
    return __uint_as_float((k & 0x80000000u) ? (k ^ 0x80000000u) : ~k);
}

__device__ __forceinline__ float wsum(float v) {
#pragma unroll
    for (int m = 16; m >= 1; m >>= 1) v += __shfl_xor(v, m, 32);
    return v;
}
__device__ __forceinline__ float dot4(float4 a, float4 b) {
    return a.x * b.x + a.y * b.y + a.z * b.z + a.w * b.w;
}

// ---------------------------------------------------------------------------
// K0: fold attn vectors through the weight matrices:
//   wl_X[h*128+k] = sum_d W_X[k,128][h*64+d] * attn_l[h,d]   (and attn_r for glob)
// ---------------------------------------------------------------------------
__global__ void k_prep_wl(const float* __restrict__ Wa, const float* __restrict__ Wb,
                          const float* __restrict__ Wg, const float* __restrict__ al,
                          const float* __restrict__ ar,
                          float* wla, float* wlb, float* wlg, float* wrg) {
    for (int idx = threadIdx.x; idx < 1024; idx += 256) {
        int m = idx >> 8, rem = idx & 255, h = rem >> 7, k = rem & 127;
        const float* W   = (m == 0) ? Wa : (m == 1) ? Wb : Wg;
        const float* att = (m == 3) ? ar : al;
        float s = 0.f;
        for (int d = 0; d < 64; ++d) s += W[k * 128 + h * 64 + d] * att[h * 64 + d];
        float* o = (m == 0) ? wla : (m == 1) ? wlb : (m == 2) ? wlg : wrg;
        o[h * 128 + k] = s;
    }
}

// ---------------------------------------------------------------------------
// K1: per-graph logits: er = u.wr, el_u = u.wl ; eu = leaky(el_u+er)
//     init m_graph = enc(eu).   wave-per-row.
// ---------------------------------------------------------------------------
__global__ void k_graph_logits(const float* __restrict__ GF, const float* __restrict__ wl,
                               const float* __restrict__ wr, float* er, float* eu_buf,
                               unsigned* m_graph, int G) {
    int wave = threadIdx.x >> 5, lane = threadIdx.x & 31;
    int row = blockIdx.x * 8 + wave;
    if (row >= G) return;
    float4 x  = ((const float4*)(GF + (size_t)row * 128))[lane];
    float4 l0 = ((const float4*)wl)[lane];
    float4 l1 = ((const float4*)(wl + 128))[lane];
    float4 r0 = ((const float4*)wr)[lane];
    float4 r1 = ((const float4*)(wr + 128))[lane];
    float pl0 = wsum(dot4(x, l0));
    float pl1 = wsum(dot4(x, l1));
    float pr0 = wsum(dot4(x, r0));
    float pr1 = wsum(dot4(x, r1));
    if (lane == 0) {
        float eu0 = leaky(pl0 + pr0), eu1 = leaky(pl1 + pr1);
        er[row * 2 + 0] = pr0;  er[row * 2 + 1] = pr1;
        eu_buf[row * 2 + 0] = eu0;  eu_buf[row * 2 + 1] = eu1;
        m_graph[row * 2 + 0] = enc_f(eu0);  m_graph[row * 2 + 1] = enc_f(eu1);
    }
}

// K2: per-atom: eh = leaky(x.wl + er[a2g[a]]); init m_atom = enc(eh)
__global__ void k_atom_logits(const float* __restrict__ AF, const float* __restrict__ wl,
                              const float* __restrict__ er, const int* __restrict__ a2g,
                              float* eh_buf, unsigned* m_atom, int N) {
    int wave = threadIdx.x >> 5, lane = threadIdx.x & 31;
    int row = blockIdx.x * 8 + wave;
    if (row >= N) return;
    float4 x  = ((const float4*)(AF + (size_t)row * 128))[lane];
    float4 l0 = ((const float4*)wl)[lane];
    float4 l1 = ((const float4*)(wl + 128))[lane];
    float p0 = wsum(dot4(x, l0));
    float p1 = wsum(dot4(x, l1));
    if (lane == 0) {
        int g = a2g[row];
        float e0 = leaky(p0 + er[g * 2 + 0]);
        float e1 = leaky(p1 + er[g * 2 + 1]);
        eh_buf[row * 2 + 0] = e0;  eh_buf[row * 2 + 1] = e1;
        m_atom[row * 2 + 0] = enc_f(e0);  m_atom[row * 2 + 1] = enc_f(e1);
    }
}

// K3: per-bond: ee = leaky(x.wl + er[a2g[dst[b]]]); atomicMax into m_atom[dst]
__global__ void k_bond_logits(const float* __restrict__ BF, const float* __restrict__ wl,
                              const float* __restrict__ er, const int* __restrict__ a2g,
                              const int* __restrict__ dst, float* ee_buf, unsigned* m_atom, int E) {
    int wave = threadIdx.x >> 5, lane = threadIdx.x & 31;
    int row = blockIdx.x * 8 + wave;
    if (row >= E) return;
    float4 x  = ((const float4*)(BF + (size_t)row * 128))[lane];
    float4 l0 = ((const float4*)wl)[lane];
    float4 l1 = ((const float4*)(wl + 128))[lane];
    float p0 = wsum(dot4(x, l0));
    float p1 = wsum(dot4(x, l1));
    if (lane == 0) {
        int a = dst[row];
        int g = a2g[a];
        float e0 = leaky(p0 + er[g * 2 + 0]);
        float e1 = leaky(p1 + er[g * 2 + 1]);
        ee_buf[row * 2 + 0] = e0;  ee_buf[row * 2 + 1] = e1;
        atomicMax(&m_atom[a * 2 + 0], enc_f(e0));
        atomicMax(&m_atom[a * 2 + 1], enc_f(e1));
    }
}

// K4: m_graph = max over atoms of m_atom (encoded-domain atomicMax)
__global__ void k_max_to_graph(const unsigned* __restrict__ m_atom, const int* __restrict__ a2g,
                               unsigned* m_graph, int N) {
    int i = blockIdx.x * 256 + threadIdx.x;
    if (i >= N * 2) return;
    int g = a2g[i >> 1];
    atomicMax(&m_graph[g * 2 + (i & 1)], m_atom[i]);
}

// K5: decode m_graph -> float (in place), eu_exp = exp(eu - m), esum = eu_exp
__global__ void k_graph_exp(unsigned* m_graph_u, float* eu_buf, float* esum, int G) {
    int i = blockIdx.x * 256 + threadIdx.x;
    if (i >= G * 2) return;
    float m = dec_f(m_graph_u[i]);
    float ex = expf(eu_buf[i] - m);
    eu_buf[i] = ex;
    esum[i] = ex;
    ((float*)m_graph_u)[i] = m;
}

__global__ void k_zero(float* p, int n) {
    int i = blockIdx.x * 256 + threadIdx.x;
    if (i < n) p[i] = 0.f;
}

// K6: eh_exp = exp(eh - m_graph); esum += eh_exp
__global__ void k_atom_exp(float* eh_buf, const float* __restrict__ m_graph_f,
                           const int* __restrict__ a2g, float* esum, int N) {
    int i = blockIdx.x * 256 + threadIdx.x;
    if (i >= N * 2) return;
    int h = i & 1, g = a2g[i >> 1];
    float ex = expf(eh_buf[i] - m_graph_f[g * 2 + h]);
    eh_buf[i] = ex;
    atomicAdd(&esum[g * 2 + h], ex);
}

// K7: ee_half = 0.5*exp(ee - m_graph); esum += ee_half  (0.5 per directed edge)
__global__ void k_bond_exp(float* ee_buf, const float* __restrict__ m_graph_f,
                           const int* __restrict__ a2g, const int* __restrict__ dst,
                           float* esum, int E) {
    int i = blockIdx.x * 256 + threadIdx.x;
    if (i >= E * 2) return;
    int h = i & 1, g = a2g[dst[i >> 1]];
    float ex = 0.5f * expf(ee_buf[i] - m_graph_f[g * 2 + h]);
    ee_buf[i] = ex;
    atomicAdd(&esum[g * 2 + h], ex);
}

// K8: alpha = numerator / esum[graph]     MODE 0: atoms, 1: bonds, 2: globals
template <int MODE>
__global__ void k_alpha_div(float* num, const float* __restrict__ esum,
                            const int* __restrict__ a2g, const int* __restrict__ dst, int n) {
    int i = blockIdx.x * 256 + threadIdx.x;
    if (i >= n * 2) return;
    int r = i >> 1, h = i & 1;
    int g = (MODE == 0) ? a2g[r] : (MODE == 1) ? a2g[dst[r]] : r;
    num[i] = num[i] / esum[g * 2 + h];
}

// ---------------------------------------------------------------------------
// K9: WMMA projection + weighted scatter-accumulate:
//   acc[g, :] += alpha[row,head] * (X @ W)[row, :]
// One wave per 16-row strip; W (128x128 f32) pre-swizzled to bf16 B-fragments
// in LDS. 8 N-tiles x 4 K-chunks of v_wmma_f32_16x16x32_bf16 per strip.
// ---------------------------------------------------------------------------
template <int MODE>
__global__ void __launch_bounds__(256)
k_wmma_proj_acc(const float* __restrict__ X, const float* __restrict__ W,
                const float* __restrict__ alpha, const int* __restrict__ a2g,
                const int* __restrict__ dst, float* __restrict__ acc, int nrows) {
    __shared__ __align__(32) __bf16 wlds[16384];  // 32 KB, [region 0..31][lane 0..31][j 0..15]

    // Pre-swizzle W into B-fragments in contiguous k order (same order as A):
    // region = nt*4 + kc; lane<16 -> N=nt*16+lane, K=kc*32+j ; lane>=16 -> K=kc*32+16+j
    for (int idx = threadIdx.x; idx < 16384; idx += 256) {
        int region = idx >> 9;
        int rem = idx & 511;
        int lane = rem >> 4;
        int j = rem & 15;
        int nt = region >> 2;
        int kc = (region & 3) << 5;
        int k = kc + ((lane >> 4) << 4) + j;
        int n = nt * 16 + (lane & 15);
        wlds[idx] = (__bf16)W[k * 128 + n];
    }
    __syncthreads();

    int wave = threadIdx.x >> 5;
    int lane = threadIdx.x & 31;
    int strip = blockIdx.x * 8 + wave;
    int row0 = strip * 16;
    if (row0 >= nrows) return;  // wave-uniform; EXEC stays all-ones for WMMA

    int lrow = lane & 15;
    int hi = lane >> 4;

    // A fragments: lane<16 holds row M=lane, K=kc..kc+15;
    // lane>=16 same row group with K=kc+16..kc+31.
    int arow = row0 + lrow;
    if (arow >= nrows) arow = nrows - 1;
    const float* xrow = X + (size_t)arow * 128;
    v16bf afrag[4];
#pragma unroll
    for (int kc = 0; kc < 4; ++kc) {
        int kbase = kc * 32 + hi * 16;
        float4 x0 = *(const float4*)(xrow + kbase);
        float4 x1 = *(const float4*)(xrow + kbase + 4);
        float4 x2 = *(const float4*)(xrow + kbase + 8);
        float4 x3 = *(const float4*)(xrow + kbase + 12);
        v16bf a;
        a[0] = (__bf16)x0.x; a[1] = (__bf16)x0.y; a[2] = (__bf16)x0.z; a[3] = (__bf16)x0.w;
        a[4] = (__bf16)x1.x; a[5] = (__bf16)x1.y; a[6] = (__bf16)x1.z; a[7] = (__bf16)x1.w;
        a[8]  = (__bf16)x2.x; a[9]  = (__bf16)x2.y; a[10] = (__bf16)x2.z; a[11] = (__bf16)x2.w;
        a[12] = (__bf16)x3.x; a[13] = (__bf16)x3.y; a[14] = (__bf16)x3.z; a[15] = (__bf16)x3.w;
        afrag[kc] = a;
    }

    // per-lane epilogue metadata: this lane's 8 C rows are M = hi*8 + r
    int grow[8];
    float alh0[8], alh1[8];
#pragma unroll
    for (int r = 0; r < 8; ++r) {
        int row = row0 + hi * 8 + r;
        if (row < nrows) {
            int g = (MODE == 0) ? a2g[row] : (MODE == 1) ? a2g[dst[row]] : row;
            grow[r] = g;
            alh0[r] = alpha[(size_t)row * 2 + 0];
            alh1[r] = alpha[(size_t)row * 2 + 1];
        } else {
            grow[r] = -1;
            alh0[r] = 0.f;
            alh1[r] = 0.f;
        }
    }

#pragma unroll
    for (int nt = 0; nt < 8; ++nt) {
        v8f c = {0.f, 0.f, 0.f, 0.f, 0.f, 0.f, 0.f, 0.f};
#pragma unroll
        for (int kc = 0; kc < 4; ++kc) {
            v16bf b = *(const v16bf*)(wlds + (nt * 4 + kc) * 512 + lane * 16);
            c = __builtin_amdgcn_wmma_f32_16x16x32_bf16(false, afrag[kc], false, b,
                                                        (short)0, c, false, false);
        }
        int col = nt * 16 + lrow;
        int head = nt >> 2;  // cols 0..63 -> head0, 64..127 -> head1
        // run-length fuse same-graph rows (a2g is sorted -> big win for atoms)
        float pend = 0.f;
        int pg = -1;
#pragma unroll
        for (int r = 0; r < 8; ++r) {
            if (grow[r] >= 0) {
                float v = c[r] * (head ? alh1[r] : alh0[r]);
                if (grow[r] == pg) {
                    pend += v;
                } else {
                    if (pg >= 0) atomicAdd(&acc[(size_t)pg * 128 + col], pend);
                    pg = grow[r];
                    pend = v;
                }
            }
        }
        if (pg >= 0) atomicAdd(&acc[(size_t)pg * 128 + col], pend);
    }
}

// K10: out[g,d] = relu(0.5*(acc[g,d] + acc[g,64+d]))   (mean over 2 heads)
__global__ void k_finalize(const float* __restrict__ acc, float* __restrict__ out, int G) {
    int i = blockIdx.x * 256 + threadIdx.x;
    if (i >= G * 64) return;
    int g = i >> 6, d = i & 63;
    float v = 0.5f * (acc[(size_t)g * 128 + d] + acc[(size_t)g * 128 + 64 + d]);
    out[i] = v > 0.f ? v : 0.f;
}

// ---------------------------------------------------------------------------
extern "C" void kernel_launch(void* const* d_in, const int* in_sizes, int n_in,
                              void* d_out, int out_size, void* d_ws, size_t ws_size,
                              hipStream_t stream) {
    const float* atom_feats   = (const float*)d_in[0];
    const float* bond_feats   = (const float*)d_in[1];
    const float* global_feats = (const float*)d_in[2];
    const float* W_atom       = (const float*)d_in[3];
    const float* W_bond       = (const float*)d_in[4];
    const float* W_glob       = (const float*)d_in[5];
    const float* attn_l       = (const float*)d_in[6];
    const float* attn_r       = (const float*)d_in[7];
    const int*   a2g          = (const int*)d_in[8];
    const int*   dst          = (const int*)d_in[9];
    float* out = (float*)d_out;

    const int N = in_sizes[0] / 128;
    const int E = in_sizes[1] / 128;
    const int G = in_sizes[2] / 128;

    char* ws = (char*)d_ws;
    auto alloc = [&](size_t bytes) {
        char* p = ws;
        ws += (bytes + 255) & ~(size_t)255;
        return p;
    };
    float*    wl_atom = (float*)alloc(256 * 4);
    float*    wl_bond = (float*)alloc(256 * 4);
    float*    wl_glob = (float*)alloc(256 * 4);
    float*    wr_glob = (float*)alloc(256 * 4);
    float*    er      = (float*)alloc((size_t)G * 2 * 4);
    float*    eu_buf  = (float*)alloc((size_t)G * 2 * 4);  // eu_raw -> eu_exp -> alpha_u
    unsigned* m_graph = (unsigned*)alloc((size_t)G * 2 * 4);  // enc uint -> decoded float
    float*    esum    = (float*)alloc((size_t)G * 2 * 4);
    float*    eh_buf  = (float*)alloc((size_t)N * 2 * 4);  // eh_raw -> eh_exp -> alpha_h
    unsigned* m_atom  = (unsigned*)alloc((size_t)N * 2 * 4);
    float*    ee_buf  = (float*)alloc((size_t)E * 2 * 4);  // ee_raw -> 0.5*ee_exp -> 0.5*alpha_e
    float*    acc     = (float*)alloc((size_t)G * 128 * 4);
    float*    m_graph_f = (float*)m_graph;

    // K0: fold attn through weights
    k_prep_wl<<<1, 256, 0, stream>>>(W_atom, W_bond, W_glob, attn_l, attn_r,
                                     wl_atom, wl_bond, wl_glob, wr_glob);
    // K1..K3: logits + encoded maxima (wave-per-row)
    k_graph_logits<<<(G + 7) / 8, 256, 0, stream>>>(global_feats, wl_glob, wr_glob,
                                                    er, eu_buf, m_graph, G);
    k_atom_logits<<<(N + 7) / 8, 256, 0, stream>>>(atom_feats, wl_atom, er, a2g,
                                                   eh_buf, m_atom, N);
    k_bond_logits<<<(E + 7) / 8, 256, 0, stream>>>(bond_feats, wl_bond, er, a2g, dst,
                                                   ee_buf, m_atom, E);
    // K4: graph max
    k_max_to_graph<<<(N * 2 + 255) / 256, 256, 0, stream>>>(m_atom, a2g, m_graph, N);
    // K5: exp on graphs, init esum; zero accumulator
    k_graph_exp<<<(G * 2 + 255) / 256, 256, 0, stream>>>(m_graph, eu_buf, esum, G);
    k_zero<<<(G * 128 + 255) / 256, 256, 0, stream>>>(acc, G * 128);
    // K6/K7: exp + softmax denominator
    k_atom_exp<<<(N * 2 + 255) / 256, 256, 0, stream>>>(eh_buf, m_graph_f, a2g, esum, N);
    k_bond_exp<<<(E * 2 + 255) / 256, 256, 0, stream>>>(ee_buf, m_graph_f, a2g, dst, esum, E);
    // K8: alphas in place
    k_alpha_div<0><<<(N * 2 + 255) / 256, 256, 0, stream>>>(eh_buf, esum, a2g, dst, N);
    k_alpha_div<1><<<(E * 2 + 255) / 256, 256, 0, stream>>>(ee_buf, esum, a2g, dst, E);
    k_alpha_div<2><<<(G * 2 + 255) / 256, 256, 0, stream>>>(eu_buf, esum, a2g, dst, G);
    // K9: WMMA projection + weighted accumulation into per-graph output
    k_wmma_proj_acc<0><<<(N + 127) / 128, 256, 0, stream>>>(atom_feats, W_atom, eh_buf,
                                                            a2g, dst, acc, N);
    k_wmma_proj_acc<1><<<(E + 127) / 128, 256, 0, stream>>>(bond_feats, W_bond, ee_buf,
                                                            a2g, dst, acc, E);
    k_wmma_proj_acc<2><<<(G + 127) / 128, 256, 0, stream>>>(global_feats, W_glob, eu_buf,
                                                            a2g, dst, acc, G);
    // K10: head mean + relu
    k_finalize<<<(G * 64 + 255) / 256, 256, 0, stream>>>(acc, out, G);
}
